// FullAttention_6983616823302
// MI455X (gfx1250) — matrix operand order repaired and checked by reading, not compiled, not run on hardware
//
#include <hip/hip_runtime.h>

// Flash attention forward (causal), fp32 in/out, f16 WMMA compute w/ fp32 accum.
// B=2, L=S=2048, H=16, E=D=64.

typedef _Float16 h16 __attribute__((ext_vector_type(16)));
typedef _Float16 h8  __attribute__((ext_vector_type(8)));
typedef _Float16 h4  __attribute__((ext_vector_type(4)));
typedef float    f8  __attribute__((ext_vector_type(8)));
typedef float    f4  __attribute__((ext_vector_type(4)));

constexpr int Bv = 2, Lv = 2048, Sv = 2048, Hv = 16, Ev = 64, Dv = 64;
constexpr int BR = 128;   // q rows per block (8 waves x 16)
constexpr int BC = 64;    // key chunk (wide: amortizes softmax fixed costs)

// Butterfly max across 16-lane groups using ds_swizzle (group-of-32 xor mode:
// offset = xor<<10 | or<<5 | and, and=0x1f). Single LDS-pipe op per step.
template <int IMM>
__device__ __forceinline__ float swz_fmax(float x) {
  int y = __builtin_amdgcn_ds_swizzle(__float_as_int(x), IMM);
  return fmaxf(x, __int_as_float(y));
}
__device__ __forceinline__ float rowmax16(float x) {
  x = swz_fmax<0x041F>(x);  // xor 1
  x = swz_fmax<0x081F>(x);  // xor 2
  x = swz_fmax<0x101F>(x);  // xor 4
  x = swz_fmax<0x201F>(x);  // xor 8
  return x;
}

__global__ __launch_bounds__(256)
void flash_attn_causal(const float* __restrict__ Q, const float* __restrict__ K,
                       const float* __restrict__ V, float* __restrict__ O)
{
  __shared__ _Float16 Kl[BC * Ev];       // K chunk  [s][e] row-major (f16)   8 KB
  __shared__ _Float16 Vt[Dv * BC];       // V chunk  [d][s'] transposed,      8 KB
                                         //   s' = (s%16)*4 + s/16 (permuted)
  __shared__ _Float16 Pl[8][16 * BC];    // per-wave P staging [m][s']       16 KB

  const int tid  = threadIdx.x;
  const int wave = tid >> 5;
  const int lane = tid & 31;
  const int n    = lane & 15;   // fragment column / A-row index
  const int g    = lane >> 4;   // lane half-group

  const int tile = blockIdx.x;
  const int h    = blockIdx.y;
  const int b    = blockIdx.z;

  const int qbase = tile * BR;
  const int qw    = qbase + wave * 16;     // first q row of this wave

  // ---- Q A-fragments, pre-scaled by 1/sqrt(E). A layout: lane holds row m =
  // lane%16, contiguous K[16g..16g+16) within each 32-slice.
  const float scale = 0.125f;
  h16 aQ[2];
  {
    const int qrow = qw + n;
    const float* qp = Q + (((b * Lv + qrow) * Hv + h) * Ev);
#pragma unroll
    for (int c = 0; c < 2; ++c) {
      f4 x0 = *(const f4*)(qp + 32 * c + 16 * g);
      f4 x1 = *(const f4*)(qp + 32 * c + 16 * g + 4);
      f4 x2 = *(const f4*)(qp + 32 * c + 16 * g + 8);
      f4 x3 = *(const f4*)(qp + 32 * c + 16 * g + 12);
      h16 a;
#pragma unroll
      for (int i = 0; i < 4; ++i) {
        a[i]      = (_Float16)(x0[i] * scale);
        a[4 + i]  = (_Float16)(x1[i] * scale);
        a[8 + i]  = (_Float16)(x2[i] * scale);
        a[12 + i] = (_Float16)(x3[i] * scale);
      }
      aQ[c] = a;
    }
  }

  // All-ones B fragment: rowsum(P) = P x ones via WMMA.
  h16 pones;
#pragma unroll
  for (int i = 0; i < 16; ++i) pones[i] = (_Float16)1.0f;

  f8 acc[4];
#pragma unroll
  for (int f = 0; f < 4; ++f) acc[f] = f8{};
  float mrow[8], lrow[8];
#pragma unroll
  for (int r = 0; r < 8; ++r) { mrow[r] = -1e30f; lrow[r] = 0.0f; }

  const int nchunk = (qbase + BR) / BC;   // = 2*tile + 2
  const int sl  = tid >> 2;                        // loader: key row (0..63)
  const int slp = (sl & 15) * 4 + (sl >> 4);       // permuted s' position
  const int e0  = (tid & 3) * 16;                  // loader: feature start

  for (int ch = 0; ch < nchunk; ++ch) {
    const int kc = ch * BC;
    __syncthreads();
    {
      const int srow = kc + sl;
      const float* kp = K + (((b * Sv + srow) * Hv + h) * Ev + e0);
      f4 a0 = *(const f4*)kp;
      f4 a1 = *(const f4*)(kp + 4);
      f4 a2 = *(const f4*)(kp + 8);
      f4 a3 = *(const f4*)(kp + 12);
      h8 hk0, hk1;
#pragma unroll
      for (int i = 0; i < 4; ++i) {
        hk0[i] = (_Float16)a0[i]; hk0[4 + i] = (_Float16)a1[i];
        hk1[i] = (_Float16)a2[i]; hk1[4 + i] = (_Float16)a3[i];
      }
      *(h8*)(&Kl[sl * Ev + e0])     = hk0;
      *(h8*)(&Kl[sl * Ev + e0 + 8]) = hk1;

      const float* vp = V + (((b * Sv + srow) * Hv + h) * Dv + e0);
      f4 b0 = *(const f4*)vp;
      f4 b1 = *(const f4*)(vp + 4);
      f4 b2 = *(const f4*)(vp + 8);
      f4 b3 = *(const f4*)(vp + 12);
#pragma unroll
      for (int i = 0; i < 4; ++i) {
        Vt[(e0 + i) * BC + slp]      = (_Float16)b0[i];
        Vt[(e0 + 4 + i) * BC + slp]  = (_Float16)b1[i];
        Vt[(e0 + 8 + i) * BC + slp]  = (_Float16)b2[i];
        Vt[(e0 + 12 + i) * BC + slp] = (_Float16)b3[i];
      }
    }
    __syncthreads();

    if (kc <= qw + 15) {   // wave-uniform: skip chunks fully above the diagonal
      // ---- S = (Q*scale) . K^T : 16 rows x 64 keys, four 16x16 C fragments
      f8 sfr[4];
#pragma unroll
      for (int t = 0; t < 4; ++t) {
        f8 c = f8{};
#pragma unroll
        for (int cc = 0; cc < 2; ++cc) {
          // B layout: lane holds column n (key 16t+n), contiguous e in [32cc+16g, +16)
          h16 bf = *(const h16*)(&Kl[(16 * t + n) * Ev + 32 * cc + 16 * g]);
          c = __builtin_amdgcn_wmma_f32_16x16x32_f16(false, aQ[cc], false, bf,
                                                     (short)0, c, false, false);
        }
        sfr[t] = c;
      }
      // ---- causal mask, only for chunks touching the diagonal
      // (C layout: row = r + 8g, col = 16t + n)
      if (kc + BC - 1 > qw) {
#pragma unroll
        for (int t = 0; t < 4; ++t)
#pragma unroll
          for (int r = 0; r < 8; ++r) {
            const int si = kc + 16 * t + n;
            const int qi = qw + r + 8 * g;
            float vv = sfr[t][r];
            sfr[t][r] = (si > qi) ? -1e30f : vv;
          }
      }
      // ---- online softmax: row max + rescale
      float alpha[8];
#pragma unroll
      for (int r = 0; r < 8; ++r) {
        float cm = fmaxf(fmaxf(sfr[0][r], sfr[1][r]), fmaxf(sfr[2][r], sfr[3][r]));
        cm = rowmax16(cm);
        const float mnew = fmaxf(mrow[r], cm);
        alpha[r] = __expf(mrow[r] - mnew);
        mrow[r]  = mnew;
#pragma unroll
        for (int t = 0; t < 4; ++t) sfr[t][r] = __expf(sfr[t][r] - mnew);
#pragma unroll
        for (int f = 0; f < 4; ++f) acc[f][r] *= alpha[r];
      }
      // ---- P: C layout -> wave-private LDS (permuted s': lane's 4 cols are
      // adjacent -> one b64 store per row, packed f16 conversion)
#pragma unroll
      for (int r = 0; r < 8; ++r) {
        h4 p;
        p[0] = (_Float16)sfr[0][r];
        p[1] = (_Float16)sfr[1][r];
        p[2] = (_Float16)sfr[2][r];
        p[3] = (_Float16)sfr[3][r];
        *(h4*)(&Pl[wave][(r + 8 * g) * BC + n * 4]) = p;
      }
      asm volatile("s_wait_dscnt 0" ::: "memory");
      h16 pa[2];
#pragma unroll
      for (int cc = 0; cc < 2; ++cc) {
        h8 plo = *(const h8*)(&Pl[wave][n * BC + 32 * cc + 16 * g]);
        h8 phi = *(const h8*)(&Pl[wave][n * BC + 32 * cc + 16 * g + 8]);
        pa[cc] = __builtin_shufflevector(plo, phi, 0, 1, 2, 3, 4, 5, 6, 7,
                                         8, 9, 10, 11, 12, 13, 14, 15);
      }
      // ---- row sums via WMMA against ones (broadcasts sum to all lanes)
      f8 ls = __builtin_amdgcn_wmma_f32_16x16x32_f16(false, pa[0], false, pones,
                                                     (short)0, f8{}, false, false);
      ls = __builtin_amdgcn_wmma_f32_16x16x32_f16(false, pa[1], false, pones,
                                                  (short)0, ls, false, false);
#pragma unroll
      for (int r = 0; r < 8; ++r) lrow[r] = lrow[r] * alpha[r] + ls[r];
      // ---- O += P . V  (4 N-tiles of 16 over D=64, 2 K=32 contractions each;
      // Vt shares the permuted s' order so the contraction pairs line up)
#pragma unroll
      for (int f = 0; f < 4; ++f) {
#pragma unroll
        for (int cc = 0; cc < 2; ++cc) {
          h16 vb = *(const h16*)(&Vt[(16 * f + n) * BC + 32 * cc + 16 * g]);
          acc[f] = __builtin_amdgcn_wmma_f32_16x16x32_f16(false, pa[cc], false, vb,
                                                          (short)0, acc[f], false, false);
        }
      }
    }
  }

  // ---- epilogue: normalize and store [B,L,H,D]
#pragma unroll
  for (int r = 0; r < 8; ++r) {
    const float inv = 1.0f / lrow[r];
    const int qi = qw + r + 8 * g;
    float* op = O + (((b * Lv + qi) * Hv + h) * Dv);
#pragma unroll
    for (int f = 0; f < 4; ++f)
      op[16 * f + n] = acc[f][r] * inv;
  }
}

extern "C" void kernel_launch(void* const* d_in, const int* in_sizes, int n_in,
                              void* d_out, int out_size, void* d_ws, size_t ws_size,
                              hipStream_t stream) {
  (void)in_sizes; (void)n_in; (void)d_ws; (void)ws_size; (void)out_size;
  const float* q = (const float*)d_in[0];
  const float* k = (const float*)d_in[1];
  const float* v = (const float*)d_in[2];
  float* out = (float*)d_out;
  dim3 grid(Lv / BR, Hv, Bv);
  dim3 block(256);
  flash_attn_causal<<<grid, block, 0, stream>>>(q, k, v, out);
}
